// GATLayer_35570919146022
// MI455X (gfx1250) — matrix operand order repaired and checked by reading, not compiled, not run on hardware
//
#include <hip/hip_runtime.h>
#include <hip/hip_bf16.h>
#include <math.h>

typedef __attribute__((ext_vector_type(16))) _Float16 v16h;
typedef __attribute__((ext_vector_type(8)))  _Float16 v8h;
typedef __attribute__((ext_vector_type(4)))  _Float16 v4h;
typedef __attribute__((ext_vector_type(8)))  float    v8f;

#define NN      4096
#define IN_DIM  256
#define HEADS   4
#define OUT_DIM 64
#define JCHUNKS 4                       // j-split for aggregation occupancy

__device__ __forceinline__ float lrelu(float v) { return v > 0.f ? v : 0.2f * v; }

// ---------------------------------------------------------------------------
// Converters: x -> f16 row-major, W -> f16 transposed (WT[col][k]) so WMMA
// fragments are contiguous vector loads instead of strided scalar chains.
// ---------------------------------------------------------------------------
__global__ void k_cvt_x(const float* __restrict__ x, _Float16* __restrict__ x16) {
    const int e = (blockIdx.x * blockDim.x + threadIdx.x) * 4;   // 1M elems
    float4 f = *(const float4*)(x + e);
    v4h p; p[0]=(_Float16)f.x; p[1]=(_Float16)f.y; p[2]=(_Float16)f.z; p[3]=(_Float16)f.w;
    *(v4h*)(x16 + e) = p;
}

__global__ void k_cvt_Wt(const float* __restrict__ W, _Float16* __restrict__ WT16) {
    const int c  = blockIdx.x;               // 256 columns
    const int k0 = threadIdx.x * 4;          // 64 threads * 4 k
    v4h p;
#pragma unroll
    for (int j = 0; j < 4; ++j) p[j] = (_Float16)W[(size_t)(k0 + j) * (HEADS*OUT_DIM) + c];
    *(v4h*)(WT16 + (size_t)c * IN_DIM + k0) = p;
}

// ---------------------------------------------------------------------------
// Kernel 1: h = x @ W via v_wmma_f32_16x16x32_f16.
// A: two 16B loads of f16 x; B: one 32B load of transposed f16 W.
// Writes h_f32 (for logits) + transposed hT_f16 [256][4096] (for aggregation).
// ---------------------------------------------------------------------------
__global__ void k_gemm_h(const _Float16* __restrict__ x16, const _Float16* __restrict__ WT16,
                         float* __restrict__ h32, _Float16* __restrict__ hT16) {
    const int lane = threadIdx.x & 31;
    const int wave = threadIdx.x >> 5;
    const int tile = blockIdx.x * 8 + wave;      // 4096 tiles
    const int tm   = tile >> 4;                  // M tile 0..255
    const int tn   = tile & 15;                  // N tile 0..15
    const int l15  = lane & 15;
    const int lhi  = lane >> 4;
    const int row  = tm * 16 + l15;
    const int col  = tn * 16 + l15;

    const _Float16* xr = x16  + (size_t)row * IN_DIM;
    const _Float16* wc = WT16 + (size_t)col * IN_DIM;

    v8f acc = {};
    for (int k = 0; k < IN_DIM; k += 32) {
        // A 16x32: lanes 0-15 K=k..k+7,k+16..k+23; lanes 16-31 K=k+16..k+23,k+32..k+39
        const int kb1 = k + lhi * 16;
        v8h g1 = *(const v8h*)(xr + kb1);
        v8h g2 = *(const v8h*)(xr + kb1 + 16);
        v16h a;
#pragma unroll
        for (int e = 0; e < 8; ++e) { a[e] = g1[e]; a[8 + e] = g2[e]; }

        // B 32x16: lanes 0-15 K=k..k+15; lanes 16-31 K=k+16..k+31
        v16h b = *(const v16h*)(wc + k + lhi * 16);

        acc = __builtin_amdgcn_wmma_f32_16x16x32_f16(false, a, false, b,
                                                     (short)0, acc, false, false);
    }

    const int rbase = tm * 16 + lhi * 8;         // D rows of this lane
    v8h pk;
#pragma unroll
    for (int r = 0; r < 8; ++r) {
        h32[(size_t)(rbase + r) * (HEADS * OUT_DIM) + col] = acc[r];
        pk[r] = (_Float16)acc[r];
    }
    *(v8h*)(hT16 + (size_t)col * NN + rbase) = pk;
}

// ---------------------------------------------------------------------------
// Kernel 2: attention logits, transposed layout es_t/ed_t [4][4096].
// ---------------------------------------------------------------------------
__global__ void k_edge(const float* __restrict__ h32, const float* __restrict__ asrc,
                       const float* __restrict__ adst,
                       float* __restrict__ es_t, float* __restrict__ ed_t) {
    const int n = blockIdx.x * blockDim.x + threadIdx.x;
    const float* hrow = h32 + (size_t)n * (HEADS * OUT_DIM);
#pragma unroll
    for (int hh = 0; hh < HEADS; ++hh) {
        float s1 = 0.f, s2 = 0.f;
        for (int d = 0; d < OUT_DIM; d += 4) {
            float4 hv = *(const float4*)(hrow + hh * OUT_DIM + d);
            float4 a1 = *(const float4*)(asrc + hh * OUT_DIM + d);
            float4 a2 = *(const float4*)(adst + hh * OUT_DIM + d);
            s1 += hv.x*a1.x + hv.y*a1.y + hv.z*a1.z + hv.w*a1.w;
            s2 += hv.x*a2.x + hv.y*a2.y + hv.z*a2.z + hv.w*a2.w;
        }
        es_t[hh * NN + n] = s1;
        ed_t[hh * NN + n] = s2;
    }
}

// ---------------------------------------------------------------------------
// Kernel 3: per-(i,head) online softmax stats (row max m, exp-sum s).
// ---------------------------------------------------------------------------
__device__ __forceinline__ void onl(int a, float ed, float es, float& m, float& s) {
    if (a != 0) {
        float sc = lrelu(es + ed);
        if (sc > m) { s = s * __expf(m - sc) + 1.0f; m = sc; }
        else        { s += __expf(sc - m); }
    }
}

__global__ void k_stats(const int* __restrict__ adj, const float* __restrict__ es_t,
                        const float* __restrict__ ed_t,
                        float* __restrict__ m_t, float* __restrict__ s_t) {
    const int i   = blockIdx.x;
    const int tid = threadIdx.x;          // 256
    const int hh  = tid >> 6;
    const int ts  = tid & 63;
    const float es = es_t[hh * NN + i];
    const int4*   arow = (const int4*)(adj + (size_t)i * NN);
    const float4* edh  = (const float4*)(ed_t + hh * NN);

    float m = -INFINITY, s = 0.0f;
    for (int j4 = ts; j4 < NN / 4; j4 += 64) {
        __builtin_prefetch(arow + j4 + 64, 0, 1);
        int4   a4 = arow[j4];
        float4 e4 = edh[j4];
        onl(a4.x, e4.x, es, m, s);
        onl(a4.y, e4.y, es, m, s);
        onl(a4.z, e4.z, es, m, s);
        onl(a4.w, e4.w, es, m, s);
    }
#pragma unroll
    for (int off = 16; off > 0; off >>= 1) {
        float mo = __shfl_xor(m, off, 32);
        float so = __shfl_xor(s, off, 32);
        float mm = fmaxf(m, mo);
        float ss = (mm == -INFINITY) ? 0.0f
                                     : (s * __expf(m - mm) + so * __expf(mo - mm));
        m = mm; s = ss;
    }
    __shared__ float lm[8], ls[8];
    const int wv = tid >> 5;
    if ((tid & 31) == 0) { lm[wv] = m; ls[wv] = s; }
    __syncthreads();
    if (tid < HEADS) {
        float m1 = lm[2 * tid], s1 = ls[2 * tid];
        float m2 = lm[2 * tid + 1], s2 = ls[2 * tid + 1];
        float mm = fmaxf(m1, m2);
        float ss = (mm == -INFINITY) ? 0.0f
                                     : (s1 * __expf(m1 - mm) + s2 * __expf(m2 - mm));
        m_t[tid * NN + i] = mm;
        s_t[tid * NN + i] = ss;
    }
}

// ---------------------------------------------------------------------------
// Kernel 4: fused exp + WMMA aggregation over a j-chunk.
// grid = (256 i-blocks, JCHUNKS). 4 waves/block, one head each, 4 accumulators
// per wave (16 rows x 64 dims). Per-head 1/(s*N) applied before the in-block
// head sum so chunk partials are purely additive.
// ---------------------------------------------------------------------------
__device__ __forceinline__ _Float16 pval(int a, float ed, float es, float mr) {
    return a ? (_Float16)__expf(lrelu(es + ed) - mr) : (_Float16)0.0f;
}

__global__ void k_aggregate(const int* __restrict__ adj, const _Float16* __restrict__ hT16,
                            const float* __restrict__ es_t, const float* __restrict__ ed_t,
                            const float* __restrict__ m_t, const float* __restrict__ s_t,
                            float* __restrict__ partial) {
    __shared__ float red[HEADS][16 * OUT_DIM];           // 16 KB
    const int i0    = blockIdx.x * 16;
    const int chunk = blockIdx.y;
    const int jlo   = chunk * (NN / JCHUNKS);
    const int jhi   = jlo + (NN / JCHUNKS);
    const int lane  = threadIdx.x & 31;
    const int head  = threadIdx.x >> 5;
    const int l15   = lane & 15;
    const int lhi   = lane >> 4;

    const int   mrow = i0 + l15;
    const float es   = es_t[head * NN + mrow];
    const float mr   = m_t[head * NN + mrow];
    const int   g1   = lhi * 16;
    const int   g2   = g1 + 8;
    const int   kb   = lhi * 16;

    const int*   arow = adj + (size_t)mrow * NN;
    const float* edh  = ed_t + head * NN;
    const _Float16* hb0 = hT16 + (size_t)(head * OUT_DIM +  0 + l15) * NN;
    const _Float16* hb1 = hT16 + (size_t)(head * OUT_DIM + 16 + l15) * NN;
    const _Float16* hb2 = hT16 + (size_t)(head * OUT_DIM + 32 + l15) * NN;
    const _Float16* hb3 = hT16 + (size_t)(head * OUT_DIM + 48 + l15) * NN;

    v8f acc0 = {}, acc1 = {}, acc2 = {}, acc3 = {};

    for (int j0 = jlo; j0 < jhi; j0 += 32) {
        __builtin_prefetch(arow + j0 + 256, 0, 1);

        int4   q0 = *(const int4*)(arow + j0 + g1);
        int4   q1 = *(const int4*)(arow + j0 + g1 + 4);
        int4   q2 = *(const int4*)(arow + j0 + g2);
        int4   q3 = *(const int4*)(arow + j0 + g2 + 4);
        float4 e0 = *(const float4*)(edh + j0 + g1);
        float4 e1 = *(const float4*)(edh + j0 + g1 + 4);
        float4 e2 = *(const float4*)(edh + j0 + g2);
        float4 e3 = *(const float4*)(edh + j0 + g2 + 4);

        v16h a;
        a[0]  = pval(q0.x, e0.x, es, mr);  a[1]  = pval(q0.y, e0.y, es, mr);
        a[2]  = pval(q0.z, e0.z, es, mr);  a[3]  = pval(q0.w, e0.w, es, mr);
        a[4]  = pval(q1.x, e1.x, es, mr);  a[5]  = pval(q1.y, e1.y, es, mr);
        a[6]  = pval(q1.z, e1.z, es, mr);  a[7]  = pval(q1.w, e1.w, es, mr);
        a[8]  = pval(q2.x, e2.x, es, mr);  a[9]  = pval(q2.y, e2.y, es, mr);
        a[10] = pval(q2.z, e2.z, es, mr);  a[11] = pval(q2.w, e2.w, es, mr);
        a[12] = pval(q3.x, e3.x, es, mr);  a[13] = pval(q3.y, e3.y, es, mr);
        a[14] = pval(q3.z, e3.z, es, mr);  a[15] = pval(q3.w, e3.w, es, mr);

        v16h b0 = *(const v16h*)(hb0 + j0 + kb);
        v16h b1 = *(const v16h*)(hb1 + j0 + kb);
        v16h b2 = *(const v16h*)(hb2 + j0 + kb);
        v16h b3 = *(const v16h*)(hb3 + j0 + kb);

        acc0 = __builtin_amdgcn_wmma_f32_16x16x32_f16(false, a, false, b0, (short)0, acc0, false, false);
        acc1 = __builtin_amdgcn_wmma_f32_16x16x32_f16(false, a, false, b1, (short)0, acc1, false, false);
        acc2 = __builtin_amdgcn_wmma_f32_16x16x32_f16(false, a, false, b2, (short)0, acc2, false, false);
        acc3 = __builtin_amdgcn_wmma_f32_16x16x32_f16(false, a, false, b3, (short)0, acc3, false, false);
    }

    // per-head 1/(s*N) scaling, then stage into LDS
    const int rbase = lhi * 8;
    float4 s0 = *(const float4*)(s_t + head * NN + i0 + rbase);
    float4 s1 = *(const float4*)(s_t + head * NN + i0 + rbase + 4);
    float scale[8];
    scale[0] = 1.0f / (s0.x * (float)NN); scale[1] = 1.0f / (s0.y * (float)NN);
    scale[2] = 1.0f / (s0.z * (float)NN); scale[3] = 1.0f / (s0.w * (float)NN);
    scale[4] = 1.0f / (s1.x * (float)NN); scale[5] = 1.0f / (s1.y * (float)NN);
    scale[6] = 1.0f / (s1.z * (float)NN); scale[7] = 1.0f / (s1.w * (float)NN);

#pragma unroll
    for (int r = 0; r < 8; ++r) {
        const int rowoff = (rbase + r) * OUT_DIM;
        red[head][rowoff +  0 + l15] = acc0[r] * scale[r];
        red[head][rowoff + 16 + l15] = acc1[r] * scale[r];
        red[head][rowoff + 32 + l15] = acc2[r] * scale[r];
        red[head][rowoff + 48 + l15] = acc3[r] * scale[r];
    }
    __syncthreads();

    // head sum -> chunk partial
    float* pout = partial + ((size_t)blockIdx.x * JCHUNKS + chunk) * (16 * OUT_DIM);
    for (int e = threadIdx.x; e < 16 * OUT_DIM; e += blockDim.x)
        pout[e] = red[0][e] + red[1][e] + red[2][e] + red[3][e];
}

// ---------------------------------------------------------------------------
// Kernel 5: sum JCHUNKS partials -> out (full overwrite, replay-safe).
// ---------------------------------------------------------------------------
__global__ void k_reduce(const float* __restrict__ partial, float* __restrict__ out) {
    const int e    = blockIdx.x * blockDim.x + threadIdx.x;    // 262144
    const int iblk = e >> 10;                                  // 16*64 per block
    const int off  = e & 1023;
    const float* p = partial + ((size_t)iblk * JCHUNKS) * 1024 + off;
    float v = 0.f;
#pragma unroll
    for (int c = 0; c < JCHUNKS; ++c) v += p[c * 1024];
    out[e] = v;
}

// ---------------------------------------------------------------------------
extern "C" void kernel_launch(void* const* d_in, const int* in_sizes, int n_in,
                              void* d_out, int out_size, void* d_ws, size_t ws_size,
                              hipStream_t stream) {
    const float* x    = (const float*)d_in[0];   // [4096,256]
    const int*   adj  = (const int*)  d_in[1];   // [4096,4096]
    const float* W    = (const float*)d_in[2];   // [256,256]
    const float* asrc = (const float*)d_in[3];   // [4,64]
    const float* adst = (const float*)d_in[4];   // [4,64]
    float*       out  = (float*)d_out;           // [4096,64]

    char* ws = (char*)d_ws;
    float*     h32     = (float*)ws;                               // 4 MB @ 0
    _Float16*  hT16    = (_Float16*)(ws + (4u  << 20));            // 2 MB
    _Float16*  x16     = (_Float16*)(ws + (6u  << 20));            // 2 MB
    _Float16*  WT16    = (_Float16*)(ws + (8u  << 20));            // 128 KB
    float*     es_t    = (float*)(ws + (8u << 20) + (256u << 10)); // 64 KB
    float*     ed_t    = es_t + NN * HEADS;
    float*     m_t     = ed_t + NN * HEADS;
    float*     s_t     = m_t  + NN * HEADS;
    float*     partial = (float*)(ws + (9u << 20));                // 4 MB

    k_cvt_x    <<<1024, 256, 0, stream>>>(x, x16);
    k_cvt_Wt   <<<IN_DIM, 64, 0, stream>>>(W, WT16);
    k_gemm_h   <<<512,  256, 0, stream>>>(x16, WT16, h32, hT16);
    k_edge     <<<16,   256, 0, stream>>>(h32, asrc, adst, es_t, ed_t);
    k_stats    <<<NN,   256, 0, stream>>>(adj, es_t, ed_t, m_t, s_t);
    k_aggregate<<<dim3(NN/16, JCHUNKS), 128, 0, stream>>>(adj, hT16, es_t, ed_t, m_t, s_t, partial);
    k_reduce   <<<1024, 256, 0, stream>>>(partial, out);
}
